// SA_Layer_42099269435599
// MI455X (gfx1250) — compile-verified, matrix operand order checked by reading, not run on hardware
//
#include <hip/hip_runtime.h>
#include <hip/hip_bf16.h>
#include <stdint.h>

// ---------------- types ----------------
typedef __attribute__((ext_vector_type(16))) __bf16 v16bf;
typedef __attribute__((ext_vector_type(8)))  float  v8f;
typedef __attribute__((ext_vector_type(4)))  uint32_t u32x4;
typedef __attribute__((ext_vector_type(8)))  int      i32x8;
typedef __attribute__((ext_vector_type(4)))  int      i32x4;

union FragA { uint4 u4[2]; v16bf v; };

#define B_BATCH   8
#define N_PTS     8192
#define NPOINT    2048
#define NSAMPLE   32
#define RADIUS    0.1f
#define NROWS     (B_BATCH*NPOINT*NSAMPLE)   // 524288
#define CIN       64
#define NTILES    (NROWS/128)                // 4096 row-tiles of 128

// ---------------- FPS: one workgroup per batch ----------------
__global__ __launch_bounds__(256) void fps_kernel(const float* __restrict__ xyz,
                                                  int* __restrict__ fps_idx) {
    const int b = blockIdx.x;
    const float* p = xyz + (size_t)b * N_PTS * 3;
    __shared__ float dist[N_PTS];
    __shared__ float smax[256];
    __shared__ int   sidx[256];
    const int t = threadIdx.x;
    for (int i = t; i < N_PTS; i += 256) dist[i] = 1e10f;
    __syncthreads();
    int far = 0;
    for (int it = 0; it < NPOINT; ++it) {
        if (t == 0) fps_idx[b * NPOINT + it] = far;
        const float cx = p[far*3+0], cy = p[far*3+1], cz = p[far*3+2];
        float best = -1.0f; int bi = 0;
        for (int i = t; i < N_PTS; i += 256) {
            float dx = p[i*3+0]-cx, dy = p[i*3+1]-cy, dz = p[i*3+2]-cz;
            float d = dx*dx + dy*dy + dz*dz;
            float dd = fminf(dist[i], d);
            dist[i] = dd;
            if (dd > best) { best = dd; bi = i; }
        }
        smax[t] = best; sidx[t] = bi;
        __syncthreads();
        for (int s = 128; s > 0; s >>= 1) {
            if (t < s) {
                if (smax[t+s] > smax[t] ||
                    (smax[t+s] == smax[t] && sidx[t+s] < sidx[t])) {
                    smax[t] = smax[t+s]; sidx[t] = sidx[t+s];
                }
            }
            __syncthreads();
        }
        far = sidx[0];
        __syncthreads();
    }
}

// ---------------- gather new_xyz ----------------
__global__ void gather_newxyz(const float* __restrict__ xyz,
                              const int* __restrict__ fps_idx,
                              float* __restrict__ out) {
    int e = blockIdx.x * 256 + threadIdx.x;       // 8*2048*3
    if (e >= B_BATCH * NPOINT * 3) return;
    int c = e % 3; int m = (e / 3) % NPOINT; int b = e / (3 * NPOINT);
    out[e] = xyz[((size_t)b * N_PTS + fps_idx[b * NPOINT + m]) * 3 + c];
}

// ---------------- ball query: one wave per query ----------------
__global__ __launch_bounds__(32) void ballquery_kernel(const float* __restrict__ xyz,
                                                       const float* __restrict__ new_xyz,
                                                       int* __restrict__ ballidx) {
    const int q = blockIdx.x;                 // 0..16383
    const int b = q >> 11, m = q & (NPOINT - 1);
    const int lane = threadIdx.x;
    __shared__ float sd[N_PTS];
    const float qx = new_xyz[(b*NPOINT+m)*3+0];
    const float qy = new_xyz[(b*NPOINT+m)*3+1];
    const float qz = new_xyz[(b*NPOINT+m)*3+2];
    const float* p = xyz + (size_t)b * N_PTS * 3;
    for (int i = lane; i < N_PTS; i += 32) {
        float dx = p[i*3+0]-qx, dy = p[i*3+1]-qy, dz = p[i*3+2]-qz;
        float d = sqrtf(fmaxf(dx*dx+dy*dy+dz*dz, 0.0f));
        sd[i] = (d <= RADIUS) ? d : 1e10f;
    }
    __syncthreads();
    unsigned long long last = 0ull;
    int firstSel = 0;
    for (int s = 0; s < NSAMPLE; ++s) {
        unsigned long long bestk = ~0ull;
        for (int i = lane; i < N_PTS; i += 32) {
            unsigned long long k =
                ((unsigned long long)__float_as_uint(sd[i]) << 32) | (unsigned)i;
            if ((s == 0 || k > last) && k < bestk) bestk = k;
        }
        #pragma unroll
        for (int off = 16; off > 0; off >>= 1) {
            unsigned long long o = __shfl_xor(bestk, off);
            if (o < bestk) bestk = o;
        }
        last = bestk;
        int idx = (int)(bestk & 0xFFFFFFFFu);
        float d = __uint_as_float((unsigned)(bestk >> 32));
        if (s == 0) firstSel = idx;
        int outIdx = (d <= 1e9f) ? idx : firstSel;   // out-of-ball -> nearest
        if (lane == 0) ballidx[q * NSAMPLE + s] = outIdx;
    }
}

// ---------------- build grouped features (bf16, row-major 524288 x 64) -----
__global__ void group_kernel(const float* __restrict__ xyz,
                             const float* __restrict__ points,
                             const float* __restrict__ new_xyz,
                             const int* __restrict__ ballidx,
                             __bf16* __restrict__ A) {
    int e = blockIdx.x * 256 + threadIdx.x;       // NROWS*64
    if (e >= NROWS * CIN) return;
    int c = e & 63; int r = e >> 6;
    int bm = r >> 5; int m = bm & (NPOINT - 1); int b = bm >> 11;
    int idx = ballidx[r];
    float v;
    if (c < 3)
        v = xyz[((size_t)b * N_PTS + idx) * 3 + c] - new_xyz[(b*NPOINT+m)*3 + c];
    else
        v = points[((size_t)b * N_PTS + idx) * 61 + (c - 3)];
    A[e] = (__bf16)v;
}

// ---------------- pack weights into WMMA B-fragment layout ----------------
// element e = ((q*NT + tile)*32 + lane)*16 + i ; pair layout mirrors A frag.
template <int COUT>
__global__ void prep_weights(const float* __restrict__ W, __bf16* __restrict__ Bp) {
    constexpr int NT = COUT / 16;
    int e = blockIdx.x * 256 + threadIdx.x;
    if (e >= 2 * NT * 32 * 16) return;
    int i    = e & 15;
    int lane = (e >> 4) & 31;
    int tile = (e >> 9) % NT;
    int q    = (e >> 9) / NT;
    int h = lane >> 4;
    int v = i >> 1;
    int kbase = (v < 4) ? (2 * v) : (2 * v + 8);
    int k = kbase + 8 * h + (i & 1);
    int c = 32 * q + k;
    int o = tile * 16 + (lane & 15);
    Bp[e] = (__bf16)W[o * CIN + c];
}

// ---------------- BN finalize: scale/shift from accumulated sums ----------
template <int COUT>
__global__ void finalize_stats(const float* __restrict__ stats,
                               const float* __restrict__ g,
                               const float* __restrict__ be,
                               float* __restrict__ scale,
                               float* __restrict__ shift) {
    int ch = threadIdx.x;
    if (ch >= COUT) return;
    const float invN = 1.0f / (float)NROWS;
    float mu  = stats[ch] * invN;
    float var = stats[COUT + ch] * invN - mu * mu;
    float inv = rsqrtf(var + 1e-5f);
    float sc = g[ch] * inv;
    scale[ch] = sc;
    shift[ch] = be[ch] - mu * sc;
}

// ---------------- TDM: DMA one 16KB row-tile of A into LDS ----------------
__device__ __forceinline__ void tdm_load_tile(const __bf16* A, int rowbase,
                                              __bf16* ldsA) {
    uint64_t ga = (uint64_t)(uintptr_t)A + (uint64_t)rowbase * (CIN * 2);
    uint32_t lds = (uint32_t)(uintptr_t)ldsA;     // low 32 bits of flat LDS addr
    u32x4 g0; i32x8 g1; i32x4 g2, g3; i32x8 g4;
    g0[0] = 1u;                                   // count=1, user descriptor
    g0[1] = lds;                                  // lds_addr
    g0[2] = (uint32_t)ga;                         // global_addr lo
    g0[3] = (uint32_t)(ga >> 32) | (2u << 30);    // global_addr hi | type=2
    g1[0] = (int)(3u << 16);                      // data_size = 8B qwords
    g1[1] = (int)((2048u & 0xFFFFu) << 16);       // tensor_dim0[15:0] = 2048
    g1[2] = (int)(1u << 16);                      // tensor_dim0 hi=0, tensor_dim1=1
    g1[3] = (int)(2048u << 16);                   // tile_dim0 = 2048 qwords (16KB)
    g1[4] = 1;                                    // tile_dim1 = 1
    g1[5] = 2048;                                 // tensor_dim0_stride
    g1[6] = 0; g1[7] = 0;
    g2[0] = g2[1] = g2[2] = g2[3] = 0;
    g3[0] = g3[1] = g3[2] = g3[3] = 0;
    g4[0] = g4[1] = g4[2] = g4[3] = 0;
    g4[4] = g4[5] = g4[6] = g4[7] = 0;
#if defined(__has_builtin)
#if __has_builtin(__builtin_amdgcn_tensor_load_to_lds)
    // clang-23 / therock form: (g0, g1, g2, g3, g4, cpol)
    __builtin_amdgcn_tensor_load_to_lds(g0, g1, g2, g3, g4, 0);
    __builtin_amdgcn_s_wait_tensorcnt(0);
#endif
#endif
}

// ---------------- fused GEMM (bf16 WMMA) with 3 epilogue modes ------------
// MODE 0: accumulate per-channel sum / sumsq -> stats (global atomics)
// MODE 1: y = relu(y*scale+shift) -> write bf16 in place (64-ch layers)
// MODE 2: y = relu(y*scale+shift), max over K=32 group -> atomicMax u32 bits
template <int COUT, int MODE>
__global__ __launch_bounds__(256) void gemm_kernel(const __bf16* A,
                                                   const __bf16* __restrict__ Bp,
                                                   float* __restrict__ stats,
                                                   const float* __restrict__ scale,
                                                   const float* __restrict__ shift,
                                                   __bf16* Aout,
                                                   unsigned* __restrict__ outMax) {
    constexpr int NT = COUT / 16;
    __shared__ __bf16 ldsA[128 * CIN];            // 16 KB staging tile
    __shared__ __bf16 ldsB[2 * NT * 32 * 16];     // packed B fragments
    __shared__ float  sStat[2 * COUT];

    const int tid  = threadIdx.x;
    const int wave = tid >> 5;
    const int lane = tid & 31;
    const int h    = lane >> 4;
    const int mrow = lane & 15;

    // stage packed weights once
    {
        const uint32_t* src = (const uint32_t*)Bp;
        uint32_t* dst = (uint32_t*)ldsB;
        for (int i = tid; i < NT * 512; i += 256) dst[i] = src[i];
    }
    if (MODE == 0)
        for (int i = tid; i < 2 * COUT; i += 256) sStat[i] = 0.0f;
    __syncthreads();

    // load B fragments to registers (reused across all row tiles)
    FragA bfr[NT][2];
    #pragma unroll
    for (int t = 0; t < NT; ++t)
        #pragma unroll
        for (int q = 0; q < 2; ++q) {
            const uint4* bp = (const uint4*)&ldsB[((q * NT + t) * 32 + lane) * 16];
            bfr[t][q].u4[0] = bp[0];
            bfr[t][q].u4[1] = bp[1];
        }

    for (int tile = blockIdx.x; tile < NTILES; tile += gridDim.x) {
        const int rowbase = tile * 128;
        __syncthreads();                           // prev tile fully consumed
        if (wave == 0) tdm_load_tile(A, rowbase, ldsA);
        __syncthreads();

        // prefetch next tile (speculative, dropped if OOB)
        if (tile + (int)gridDim.x < NTILES)
            __builtin_prefetch(A + (size_t)(tile + gridDim.x) * 128 * CIN, 0, 3);

        const int r0 = wave * 16;
        const __bf16* rp = &ldsA[(r0 + mrow) * CIN];
        FragA a0, a1;
        a0.u4[0] = *(const uint4*)(rp +      8 * h);   // K 0..7 / 8..15
        a0.u4[1] = *(const uint4*)(rp + 16 + 8 * h);   // K16..23 / 24..31
        a1.u4[0] = *(const uint4*)(rp + 32 + 8 * h);   // K32..39 / 40..47
        a1.u4[1] = *(const uint4*)(rp + 48 + 8 * h);   // K48..55 / 56..63

        v8f acc[NT];
        #pragma unroll
        for (int t = 0; t < NT; ++t) {
            v8f c = {};
            c = __builtin_amdgcn_wmma_f32_16x16x32_bf16(false, a0.v, false,
                                                        bfr[t][0].v, (short)0, c,
                                                        false, false);
            c = __builtin_amdgcn_wmma_f32_16x16x32_bf16(false, a1.v, false,
                                                        bfr[t][1].v, (short)0, c,
                                                        false, false);
            acc[t] = c;
        }

        if (MODE == 0) {
            #pragma unroll
            for (int t = 0; t < NT; ++t) {
                float s = 0.f, s2 = 0.f;
                #pragma unroll
                for (int j = 0; j < 8; ++j) { float y = acc[t][j]; s += y; s2 += y * y; }
                s  += __shfl_xor(s, 16);
                s2 += __shfl_xor(s2, 16);
                if (lane < 16) {
                    atomicAdd(&sStat[t * 16 + lane], s);
                    atomicAdd(&sStat[COUT + t * 16 + lane], s2);
                }
            }
        } else if (MODE == 1) {
            #pragma unroll
            for (int t = 0; t < NT; ++t) {
                const int ch = t * 16 + mrow;
                const float sc = scale[ch], sh = shift[ch];
                #pragma unroll
                for (int j = 0; j < 8; ++j) {
                    float y = fmaxf(acc[t][j] * sc + sh, 0.0f);
                    int row = rowbase + r0 + h * 8 + j;
                    Aout[(size_t)row * CIN + ch] = (__bf16)y;
                }
            }
        } else {
            const int grp = (rowbase + r0) >> 5;   // 16-row wave tile fits a group
            #pragma unroll
            for (int t = 0; t < NT; ++t) {
                const int ch = t * 16 + mrow;
                const float sc = scale[ch], sh = shift[ch];
                float mx = 0.0f;
                #pragma unroll
                for (int j = 0; j < 8; ++j)
                    mx = fmaxf(mx, fmaxf(acc[t][j] * sc + sh, 0.0f));
                mx = fmaxf(mx, __shfl_xor(mx, 16));
                if (lane < 16)
                    atomicMax(&outMax[(size_t)grp * COUT + ch], __float_as_uint(mx));
            }
        }
    }

    if (MODE == 0) {
        __syncthreads();
        for (int i = tid; i < 2 * COUT; i += 256) atomicAdd(&stats[i], sStat[i]);
    }
}

// ---------------- host side ----------------
extern "C" void kernel_launch(void* const* d_in, const int* in_sizes, int n_in,
                              void* d_out, int out_size, void* d_ws, size_t ws_size,
                              hipStream_t stream) {
    (void)in_sizes; (void)n_in; (void)ws_size;
    const float* xyz    = (const float*)d_in[0];
    const float* points = (const float*)d_in[1];
    const float* W0 = (const float*)d_in[2];
    const float* g0 = (const float*)d_in[4];
    const float* be0 = (const float*)d_in[5];
    const float* W1 = (const float*)d_in[6];
    const float* g1 = (const float*)d_in[8];
    const float* be1 = (const float*)d_in[9];
    const float* W2 = (const float*)d_in[10];
    const float* g2 = (const float*)d_in[12];
    const float* be2 = (const float*)d_in[13];

    float* fout = (float*)d_out;                       // [49152 new_xyz][2097152 new_points]
    char* ws = (char*)d_ws;
    int*    fps_idx = (int*)(ws + 0);                  // 64 KB
    int*    ballidx = (int*)(ws + 65536);              // 2 MB
    __bf16* Bp0     = (__bf16*)(ws + 2162688);         // 8 KB
    __bf16* Bp1     = (__bf16*)(ws + 2170880);         // 8 KB
    __bf16* Bp2     = (__bf16*)(ws + 2179072);         // 16 KB
    float*  stats   = (float*)(ws + 2195456);          // 1 KB
    float*  scale   = (float*)(ws + 2196480);
    float*  shift   = (float*)(ws + 2196992);
    __bf16* A       = (__bf16*)(ws + 2228224);         // 64 MB (524288 x 64 bf16)

    (void)hipMemsetAsync(d_out, 0, (size_t)out_size * sizeof(float), stream);

    fps_kernel<<<B_BATCH, 256, 0, stream>>>(xyz, fps_idx);
    gather_newxyz<<<(B_BATCH*NPOINT*3 + 255)/256, 256, 0, stream>>>(xyz, fps_idx, fout);
    ballquery_kernel<<<B_BATCH*NPOINT, 32, 0, stream>>>(xyz, fout, ballidx);
    group_kernel<<<(NROWS*CIN + 255)/256, 256, 0, stream>>>(xyz, points, fout, ballidx, A);

    prep_weights<64><<<(4096 + 255)/256, 256, 0, stream>>>(W0, Bp0);
    prep_weights<64><<<(4096 + 255)/256, 256, 0, stream>>>(W1, Bp1);
    prep_weights<128><<<(8192 + 255)/256, 256, 0, stream>>>(W2, Bp2);

    // layer 0 (64 -> 64)
    (void)hipMemsetAsync(stats, 0, 2 * 64 * sizeof(float), stream);
    gemm_kernel<64, 0><<<512, 256, 0, stream>>>(A, Bp0, stats, nullptr, nullptr, nullptr, nullptr);
    finalize_stats<64><<<1, 64, 0, stream>>>(stats, g0, be0, scale, shift);
    gemm_kernel<64, 1><<<512, 256, 0, stream>>>(A, Bp0, nullptr, scale, shift, A, nullptr);

    // layer 1 (64 -> 64)
    (void)hipMemsetAsync(stats, 0, 2 * 64 * sizeof(float), stream);
    gemm_kernel<64, 0><<<512, 256, 0, stream>>>(A, Bp1, stats, nullptr, nullptr, nullptr, nullptr);
    finalize_stats<64><<<1, 64, 0, stream>>>(stats, g1, be1, scale, shift);
    gemm_kernel<64, 1><<<512, 256, 0, stream>>>(A, Bp1, nullptr, scale, shift, A, nullptr);

    // layer 2 (64 -> 128) + fused max-pool over K into new_points
    (void)hipMemsetAsync(stats, 0, 2 * 128 * sizeof(float), stream);
    gemm_kernel<128, 0><<<512, 256, 0, stream>>>(A, Bp2, stats, nullptr, nullptr, nullptr, nullptr);
    finalize_stats<128><<<1, 128, 0, stream>>>(stats, g2, be2, scale, shift);
    gemm_kernel<128, 2><<<512, 256, 0, stream>>>(A, Bp2, nullptr, scale, shift, nullptr,
                                                 (unsigned*)(fout + B_BATCH*NPOINT*3));
}
